// OriVAEDecoder_30013231464959
// MI455X (gfx1250) — matrix operand order repaired and checked by reading, not compiled, not run on hardware
//
#include <hip/hip_runtime.h>

typedef __bf16 bf16;
typedef __attribute__((ext_vector_type(8)))  bf16  bf16x8;
typedef __attribute__((ext_vector_type(16))) bf16  v16bf;
typedef __attribute__((ext_vector_type(8)))  float v8f;

// ------------------------------------------------------------
// CDNA5 async global->LDS copy (16B per lane), tracked by ASYNCcnt.
// lds_off: byte offset within workgroup LDS (low 32 bits of flat addr).
// ------------------------------------------------------------
__device__ __forceinline__ void async_copy16(unsigned lds_off, const bf16* g)
{
    unsigned long long ga = (unsigned long long)g;
    asm volatile("global_load_async_to_lds_b128 %0, %1, off"
                 :: "v"(lds_off), "v"(ga) : "memory");
}

__device__ __forceinline__ void wait_async0()
{
    asm volatile("s_wait_asynccnt 0x0" ::: "memory");
}

// ============================================================
// Full-tile WMMA GEMM (N % 64 == 0), bf16 out + ReLU.
// Block = 256 thr = 8 waves. Block tile: 128 M x 64 N.
// B slab (32K x 64N bf16 = 4KB) staged to LDS via async copies,
// double buffered, shared by all 8 waves. A loaded per-wave.
// M % 16 == 0, K % 32 == 0 required.
// ============================================================
__global__ __launch_bounds__(256) void gemm_wmma_lds(
    const bf16* __restrict__ A, int lda,
    const bf16* __restrict__ Wt, const float* __restrict__ bias,
    bf16* __restrict__ C, int ldc, int M, int N, int K)
{
    __shared__ __align__(128) bf16 sB[2][32 * 64]; // [buf][n_local*32 + k_local]

    const int tid   = threadIdx.x;
    const int lane  = tid & 31;
    const int wave  = tid >> 5;
    const int lcol  = lane & 15;
    const int lhalf = lane >> 4;

    const int Nt64 = N >> 6;
    const int Mt16 = M >> 4;
    const int bm   = blockIdx.x / Nt64;
    const int tn   = blockIdx.x - bm * Nt64;
    const int tm   = bm * 8 + wave;
    const bool valid = tm < Mt16;
    const int tmc  = valid ? tm : (Mt16 - 1);   // clamped: still computes, store guarded

    // staging role: each thread copies one 16B chunk of the 4KB B slab
    const int srow   = tid >> 2;   // 0..63  (n within tile)
    const int schunk = tid & 3;    // 0..3   (16B chunk within 64B K-row)
    const bf16* gB = Wt + (size_t)(tn * 64 + srow) * K + schunk * 8;
    const unsigned ldsOff0 =
        (unsigned)(size_t)(void*)&sB[0][srow * 32 + schunk * 8];
    const unsigned ldsOff1 =
        (unsigned)(size_t)(void*)&sB[1][srow * 32 + schunk * 8];

    // A row base: row (tmc*16 + lcol), K-offset lhalf*16
    const bf16* arow = A + (size_t)(tmc * 16 + lcol) * lda + lhalf * 16;

    v8f acc[4] = {};

    // prologue: stage K-slab 0 into buffer 0
    async_copy16(ldsOff0, gB);

    const int nsteps = K >> 5;
    for (int s = 0; s < nsteps; ++s) {
        const int kt = s << 5;
        const int p  = s & 1;

        wait_async0();        // my async copies (buffer p) landed in LDS
        __syncthreads();      // everyone's copies landed; prev readers of
                              // buffer (1-p) are also past their reads

        if (s + 1 < nsteps)   // overlap: stage next slab into other buffer
            async_copy16(p ? ldsOff0 : ldsOff1, gB + (size_t)(kt + 32));

        if (kt + 32 < K) __builtin_prefetch(arow + kt + 32, 0, 3);

        // A 16x32 bf16: elems 0..15 -> K = kt+lhalf*16..kt+lhalf*16+15
        bf16x8 a0 = *(const bf16x8*)(arow + kt);
        bf16x8 a1 = *(const bf16x8*)(arow + kt + 8);
        v16bf va;
#pragma unroll
        for (int i = 0; i < 8; ++i) { va[i] = a0[i]; va[i + 8] = a1[i]; }

        const bf16* sb = &sB[p][0];
#pragma unroll
        for (int j = 0; j < 4; ++j) {
            const int nl = j * 16 + lcol;
            // B 32x16: lanes 0-15 -> K kt..kt+15, lanes 16-31 -> kt+16..kt+31
            v16bf vb = *(const v16bf*)(sb + nl * 32 + lhalf * 16);
            acc[j] = __builtin_amdgcn_wmma_f32_16x16x32_bf16(
                false, va, false, vb, (short)0, acc[j], false, false);
        }
    }

    if (!valid) return;
#pragma unroll
    for (int j = 0; j < 4; ++j) {
        const int n = tn * 64 + j * 16 + lcol;
        const float bv = bias[n];
#pragma unroll
        for (int r = 0; r < 8; ++r) {
            const int m = tm * 16 + lhalf * 8 + r;
            float v = acc[j][r] + bv;
            v = v > 0.0f ? v : 0.0f;
            C[(size_t)m * ldc + n] = (bf16)v;
        }
    }
}

// ============================================================
// Edge WMMA GEMM (small / non-multiple-of-64 N), register-direct B.
// Used for the N=6 and N=24 heads (fp32 out, no ReLU).
// ============================================================
template <int RELU, int OUTBF16>
__global__ __launch_bounds__(256) void gemm_wmma_edge(
    const bf16* __restrict__ A, int lda,
    const bf16* __restrict__ Wt, const float* __restrict__ bias,
    void* __restrict__ Cout, int ldc, int M, int N, int K)
{
    const int lane  = threadIdx.x & 31;
    const int wave  = threadIdx.x >> 5;
    const int lcol  = lane & 15;
    const int lhalf = lane >> 4;

    const int Nt64 = (N + 63) >> 6;
    const int Mt16 = M >> 4;
    int gid = blockIdx.x * 8 + wave;
    int tm  = gid / Nt64;
    int tn  = gid - tm * Nt64;
    if (tm >= Mt16) return;

    v8f acc[4] = {};
    const bf16* arow = A + (size_t)(tm * 16 + lcol) * lda + lhalf * 16;

    for (int kt = 0; kt < K; kt += 32) {
        bf16x8 a0 = *(const bf16x8*)(arow + kt);
        bf16x8 a1 = *(const bf16x8*)(arow + kt + 8);
        v16bf va;
#pragma unroll
        for (int i = 0; i < 8; ++i) { va[i] = a0[i]; va[i + 8] = a1[i]; }

#pragma unroll
        for (int j = 0; j < 4; ++j) {
            int n = tn * 64 + j * 16 + lcol;
            v16bf vb = {};
            if (n < N)
                vb = *(const v16bf*)(Wt + (size_t)n * K + kt + lhalf * 16);
            acc[j] = __builtin_amdgcn_wmma_f32_16x16x32_bf16(
                false, va, false, vb, (short)0, acc[j], false, false);
        }
    }

#pragma unroll
    for (int j = 0; j < 4; ++j) {
        int n = tn * 64 + j * 16 + lcol;
        if (n >= N) continue;
        float bv = bias[n];
#pragma unroll
        for (int r = 0; r < 8; ++r) {
            int m = tm * 16 + lhalf * 8 + r;
            float v = acc[j][r] + bv;
            if (RELU) v = v > 0.0f ? v : 0.0f;
            if (OUTBF16) ((bf16*)Cout)[(size_t)m * ldc + n] = (bf16)v;
            else         ((float*)Cout)[(size_t)m * ldc + n] = v;
        }
    }
}

// ============================================================
// Support kernels
// ============================================================
__global__ __launch_bounds__(256) void fill_zero_f(float* __restrict__ p, int n)
{
    int i = blockIdx.x * 256 + threadIdx.x;
    if (i < n) p[i] = 0.0f;
}

// dst[N x K](bf16) = transpose(src[K x N](f32))
__global__ __launch_bounds__(256) void transpose_w_bf16(
    const float* __restrict__ src, bf16* __restrict__ dst, int K, int N)
{
    int idx = blockIdx.x * 256 + threadIdx.x;
    if (idx >= K * N) return;
    int n = idx / K, k = idx - n * K;
    dst[idx] = (bf16)src[(size_t)k * N + n];
}

// obj_vecs = [obj_emb[objs] (96) | attr_emb[attrs] (32)]; also keeps attrv
__global__ __launch_bounds__(256) void embed_objs_k(
    const float* __restrict__ obj_emb, const float* __restrict__ attr_emb,
    const int* __restrict__ objs, const int* __restrict__ attrs,
    bf16* __restrict__ objv, bf16* __restrict__ attrv, int O)
{
    int idx = blockIdx.x * 256 + threadIdx.x;
    if (idx >= O * 128) return;
    int i = idx >> 7, c = idx & 127;
    float v;
    if (c < 96) {
        v = obj_emb[(size_t)objs[i] * 96 + c];
    } else {
        int a = c - 96;
        v = attr_emb[(size_t)attrs[i] * 32 + a];
        attrv[(size_t)i * 32 + a] = (bf16)v;
    }
    objv[idx] = (bf16)v;
}

__global__ __launch_bounds__(256) void embed_preds_k(
    const float* __restrict__ pred_emb, const int* __restrict__ triples,
    bf16* __restrict__ predv, int T)
{
    int idx = blockIdx.x * 256 + threadIdx.x;
    if (idx >= T * 128) return;
    int t = idx >> 7, c = idx & 127;
    int p = triples[3 * t + 1];
    predv[idx] = (bf16)pred_emb[(size_t)p * 128 + c];
}

// cur_t = [obj_vecs[s] | pred_vecs | obj_vecs[o]]  (T x 384 bf16)
__global__ __launch_bounds__(256) void gather_triples_k(
    const int* __restrict__ triples, const bf16* __restrict__ objv,
    const bf16* __restrict__ predv, bf16* __restrict__ curT, int T)
{
    int idx = blockIdx.x * 256 + threadIdx.x;
    if (idx >= T * 384) return;
    int t = idx / 384, c = idx - t * 384;
    bf16 v;
    if (c < 128)       v = objv[(size_t)triples[3 * t] * 128 + c];
    else if (c < 256)  v = predv[(size_t)t * 128 + (c - 128)];
    else               v = objv[(size_t)triples[3 * t + 2] * 128 + (c - 256)];
    curT[idx] = v;
}

// pred_vecs = new_t[:, 512:640]
__global__ __launch_bounds__(256) void update_pred_k(
    const bf16* __restrict__ newT, bf16* __restrict__ predv, int T)
{
    int idx = blockIdx.x * 256 + threadIdx.x;
    if (idx >= T * 128) return;
    int t = idx >> 7, j = idx & 127;
    predv[idx] = newT[(size_t)t * 1152 + 512 + j];
}

// scatter-add pooling (fp32 atomics) + counts
__global__ __launch_bounds__(256) void scatter_pool_k(
    const int* __restrict__ triples, const bf16* __restrict__ newT,
    float* __restrict__ pooled, float* __restrict__ counts, int T)
{
    int t = blockIdx.x;
    if (t >= T) return;
    int s = triples[3 * t], o = triples[3 * t + 2];
    const bf16* row = newT + (size_t)t * 1152;
#pragma unroll
    for (int q = 0; q < 2; ++q) {
        int j = threadIdx.x + q * 256;
        atomicAdd(&pooled[(size_t)s * 512 + j], (float)row[j]);        // new_s
        atomicAdd(&pooled[(size_t)o * 512 + j], (float)row[640 + j]);  // new_o
    }
    if (threadIdx.x == 0) {
        atomicAdd(&counts[s], 1.0f);
        atomicAdd(&counts[o], 1.0f);
    }
}

__global__ __launch_bounds__(256) void div_conv_k(
    const float* __restrict__ pooled, const float* __restrict__ counts,
    bf16* __restrict__ pooledb, int O)
{
    int idx = blockIdx.x * 256 + threadIdx.x;
    if (idx >= O * 512) return;
    int i = idx >> 9;
    float c = counts[i];
    c = c < 1.0f ? 1.0f : c;
    pooledb[idx] = (bf16)(pooled[idx] / c);
}

// box_in = [obj_vecs_final (128) | z (128) | attr (32)] as bf16 (O x 288)
__global__ __launch_bounds__(256) void build_boxin_k(
    const bf16* __restrict__ objv, const float* __restrict__ z,
    const bf16* __restrict__ attrv, bf16* __restrict__ boxin, int O)
{
    int idx = blockIdx.x * 256 + threadIdx.x;
    if (idx >= O * 288) return;
    int i = idx / 288, c = idx - i * 288;
    bf16 v;
    if (c < 128)      v = objv[(size_t)i * 128 + c];
    else if (c < 256) v = (bf16)z[(size_t)i * 128 + (c - 128)];
    else              v = attrv[(size_t)i * 32 + (c - 256)];
    boxin[idx] = v;
}

__global__ __launch_bounds__(256) void log_softmax_k(
    const float* __restrict__ logits, float* __restrict__ out, int O)
{
    int i = blockIdx.x * 256 + threadIdx.x;
    if (i >= O) return;
    const float* x = logits + (size_t)i * 24;
    float m = x[0];
#pragma unroll
    for (int j = 1; j < 24; ++j) m = x[j] > m ? x[j] : m;
    float s = 0.0f;
#pragma unroll
    for (int j = 0; j < 24; ++j) s += __expf(x[j] - m);
    float l = __logf(s);
#pragma unroll
    for (int j = 0; j < 24; ++j) out[(size_t)i * 24 + j] = x[j] - m - l;
}

// ============================================================
// Host side
// ============================================================
static inline int ceil_div(long a, int b) { return (int)((a + b - 1) / b); }

static inline void launch_gemm(const bf16* A, int lda, const bf16* Wt,
                               const float* bias, void* C, int ldc,
                               int M, int N, int K, bool relu, bool obf,
                               hipStream_t s)
{
    if ((N & 63) == 0 && relu && obf) {
        int Mb = (M / 16 + 7) / 8;
        int blocks = Mb * (N / 64);
        gemm_wmma_lds<<<blocks, 256, 0, s>>>(A, lda, Wt, bias, (bf16*)C,
                                             ldc, M, N, K);
    } else {
        int Nt64 = (N + 63) / 64;
        long waves = (long)(M / 16) * Nt64;
        int blocks = (int)((waves + 7) / 8);
        if (!relu && !obf)
            gemm_wmma_edge<0, 0><<<blocks, 256, 0, s>>>(A, lda, Wt, bias, C,
                                                        ldc, M, N, K);
        else
            gemm_wmma_edge<1, 1><<<blocks, 256, 0, s>>>(A, lda, Wt, bias, C,
                                                        ldc, M, N, K);
    }
}

extern "C" void kernel_launch(void* const* d_in, const int* in_sizes, int n_in,
                              void* d_out, int out_size, void* d_ws, size_t ws_size,
                              hipStream_t stream)
{
    (void)n_in; (void)out_size; (void)ws_size;

    // ---- inputs (jax pytree flat order: data arrays then sorted-key params) ----
    const float* z        = (const float*)d_in[0];
    const int*   objs     = (const int*)d_in[1];
    const int*   triples  = (const int*)d_in[2];
    const int*   attrs    = (const int*)d_in[3];
    const float* angle0_b = (const float*)d_in[4];
    const float* angle0_w = (const float*)d_in[5];
    const float* angle1_b = (const float*)d_in[6];
    const float* angle1_w = (const float*)d_in[7];
    const float* attr_emb = (const float*)d_in[8];
    const float* box0_b   = (const float*)d_in[9];
    const float* box0_w   = (const float*)d_in[10];
    const float* box1_b   = (const float*)d_in[11];
    const float* box1_w   = (const float*)d_in[12];
    // gconv layers: d_in[13 + 8*l + {0:n1_0_b,1:n1_0_w,2:n1_1_b,3:n1_1_w,
    //                                4:n2_0_b,5:n2_0_w,6:n2_1_b,7:n2_1_w}]
    const float* obj_emb  = (const float*)d_in[53];
    const float* pred_emb = (const float*)d_in[54];

    const int O = in_sizes[1];        // 20000
    const int T = in_sizes[2] / 3;    // 50000

    // ---- workspace carve ----
    char* wsb = (char*)d_ws;
    size_t off = 0;
    auto carve = [&](size_t bytes) -> char* {
        char* p = wsb + off;
        off = (off + bytes + 255) & ~(size_t)255;
        return p;
    };

    // bf16 transposed weights (24 matrices)
    struct WSpec { const float* src; int K; int N; };
    WSpec specs[24];
    int wi = 0;
    for (int l = 0; l < 5; ++l) {
        int base = 13 + 8 * l;
        specs[wi++] = { (const float*)d_in[base + 1], 384, 512 };   // net1[0].w
        specs[wi++] = { (const float*)d_in[base + 3], 512, 1152 };  // net1[1].w
        specs[wi++] = { (const float*)d_in[base + 5], 512, 512 };   // net2[0].w
        specs[wi++] = { (const float*)d_in[base + 7], 512, 128 };   // net2[1].w
    }
    specs[wi++] = { box0_w,   288, 512 };
    specs[wi++] = { box1_w,   512, 6   };
    specs[wi++] = { angle0_w, 256, 512 };
    specs[wi++] = { angle1_w, 512, 24  };

    bf16* wt_ptr[24];
    for (int i = 0; i < 24; ++i)
        wt_ptr[i] = (bf16*)carve((size_t)specs[i].K * specs[i].N * sizeof(bf16));

    bf16*  objv    = (bf16*)carve((size_t)O * 128 * sizeof(bf16));
    bf16*  attrv   = (bf16*)carve((size_t)O * 32  * sizeof(bf16));
    bf16*  predv   = (bf16*)carve((size_t)T * 128 * sizeof(bf16));
    bf16*  curT    = (bf16*)carve((size_t)T * 384 * sizeof(bf16));
    bf16*  h1      = (bf16*)carve((size_t)T * 512 * sizeof(bf16));
    bf16*  newT    = (bf16*)carve((size_t)T * 1152 * sizeof(bf16));
    float* pooled  = (float*)carve((size_t)O * 512 * sizeof(float));
    float* counts  = (float*)carve((size_t)O * sizeof(float));
    bf16*  pooledb = (bf16*)carve((size_t)O * 512 * sizeof(bf16));
    // Phase-2 buffers alias the newT region (free after last scatter):
    bf16*  boxin     = (bf16*)((char*)newT);
    bf16*  hbox      = (bf16*)((char*)newT + (size_t)O * 288 * sizeof(bf16));
    bf16*  hang      = (bf16*)((char*)hbox + (size_t)O * 512 * sizeof(bf16));
    float* anglogits = (float*)((char*)hang + (size_t)O * 512 * sizeof(bf16));
    bf16*  h2 = h1; // reuse (h1 dead once needed)

    // ---- 1) convert + transpose weights to bf16 Wt[N][K] ----
    for (int i = 0; i < 24; ++i) {
        long n = (long)specs[i].K * specs[i].N;
        transpose_w_bf16<<<ceil_div(n, 256), 256, 0, stream>>>(
            specs[i].src, wt_ptr[i], specs[i].K, specs[i].N);
    }

    // ---- 2) embeddings ----
    embed_objs_k<<<ceil_div((long)O * 128, 256), 256, 0, stream>>>(
        obj_emb, attr_emb, objs, attrs, objv, attrv, O);
    embed_preds_k<<<ceil_div((long)T * 128, 256), 256, 0, stream>>>(
        pred_emb, triples, predv, T);

    // ---- 3) graph-conv layers ----
    for (int l = 0; l < 5; ++l) {
        int base = 13 + 8 * l;
        const float* b1a = (const float*)d_in[base + 0];
        const float* b1b = (const float*)d_in[base + 2];
        const float* b2a = (const float*)d_in[base + 4];
        const float* b2b = (const float*)d_in[base + 6];
        bf16* w1a = wt_ptr[4 * l + 0];
        bf16* w1b = wt_ptr[4 * l + 1];
        bf16* w2a = wt_ptr[4 * l + 2];
        bf16* w2b = wt_ptr[4 * l + 3];

        gather_triples_k<<<ceil_div((long)T * 384, 256), 256, 0, stream>>>(
            triples, objv, predv, curT, T);

        launch_gemm(curT, 384, w1a, b1a, h1, 512, T, 512, 384, true, true, stream);
        launch_gemm(h1, 512, w1b, b1b, newT, 1152, T, 1152, 512, true, true, stream);

        update_pred_k<<<ceil_div((long)T * 128, 256), 256, 0, stream>>>(newT, predv, T);

        fill_zero_f<<<ceil_div((long)O * 512, 256), 256, 0, stream>>>(pooled, O * 512);
        fill_zero_f<<<ceil_div((long)O, 256), 256, 0, stream>>>(counts, O);
        scatter_pool_k<<<T, 256, 0, stream>>>(triples, newT, pooled, counts, T);
        div_conv_k<<<ceil_div((long)O * 512, 256), 256, 0, stream>>>(
            pooled, counts, pooledb, O);

        launch_gemm(pooledb, 512, w2a, b2a, h2, 512, O, 512, 512, true, true, stream);
        launch_gemm(h2, 512, w2b, b2b, objv, 128, O, 128, 512, true, true, stream);
    }

    // ---- 4) heads ----
    build_boxin_k<<<ceil_div((long)O * 288, 256), 256, 0, stream>>>(
        objv, z, attrv, boxin, O);

    // boxes: relu(box_in @ W0 + b0) @ W1 + b1  -> d_out[0 .. O*6)
    launch_gemm(boxin, 288, wt_ptr[20], box0_b, hbox, 512, O, 512, 288, true, true, stream);
    launch_gemm(hbox, 512, wt_ptr[21], box1_b, d_out, 6, O, 6, 512, false, false, stream);

    // angles: log_softmax(relu(obj_cat @ W0 + b0) @ W1 + b1) -> d_out[O*6 ..)
    // obj_cat = first 256 cols of box_in (lda = 288)
    launch_gemm(boxin, 288, wt_ptr[22], angle0_b, hang, 512, O, 512, 256, true, true, stream);
    launch_gemm(hang, 512, wt_ptr[23], angle1_b, anglogits, 24, O, 24, 512, false, false, stream);

    float* ang_out = (float*)d_out + (size_t)O * 6;
    log_softmax_k<<<ceil_div((long)O, 256), 256, 0, stream>>>(anglogits, ang_out, O);
}
